// TransitionModel_30640296690324
// MI455X (gfx1250) — compile-verified, matrix operand order checked by reading, not run on hardware
//
#include <hip/hip_runtime.h>
#include <hip/hip_bf16.h>
#include <math.h>

typedef __attribute__((ext_vector_type(16))) _Float16 v16h;
typedef __attribute__((ext_vector_type(8)))  _Float16 v8h;
typedef __attribute__((ext_vector_type(8)))  float    v8f;

#define B_     512
#define T_     48
#define STATE_ 256
#define ACT_   64
#define BEL_   1024
#define HID_   1024
#define EMB_   1024

// ---------------- WMMA GEMM: out = act(A[M,K] @ Wt^T + bias) ----------------
// A f16 row-major [M,K]; Wt f16 row-major [N,K] (pre-transposed weights).
// BM=BN=128, BK=32. 256 threads = 8 wave32s, each wave owns 64x32 = 4x2 tiles.
// Tiles staged via gfx1250 async global->LDS copies; software-pipelined with
// two statically-indexed LDS buffers (K/BK is even for every GEMM we launch).
#define BM  128
#define BN  128
#define BK  32
#define LDT 40   // padded LDS row stride in halves (80B rows, bank-conflict-free)

__device__ __forceinline__ void async_b128(unsigned lds_off, const _Float16* gp)
{
    asm volatile("global_load_async_to_lds_b128 %0, %1, off"
                 :: "v"(lds_off), "v"(gp) : "memory");
}
__device__ __forceinline__ void wait_async_4()
{
    asm volatile("s_wait_asynccnt 0x4" ::: "memory");
}
__device__ __forceinline__ void wait_async_0()
{
    asm volatile("s_wait_asynccnt 0x0" ::: "memory");
}

__global__ __launch_bounds__(256)
void gemm_f16_wmma(const _Float16* __restrict__ A,
                   const _Float16* __restrict__ Wt,   // [N,K]
                   const float*    __restrict__ bias,
                   float*          __restrict__ outF, // nullable
                   _Float16*       __restrict__ outH, // nullable
                   int M, int K, int N, int relu)
{
    // [buf][A|B][128*40]
    __shared__ __align__(16) _Float16 smem[2][2][BM * LDT];

    const int tid  = threadIdx.x;
    const int lane = tid & 31;
    const int wave = tid >> 5;
    const int wm   = wave >> 2;   // 0..1 : 64-row band
    const int wn   = wave & 3;    // 0..3 : 32-col band

    const int rowBase = blockIdx.y * BM;
    const int colBase = blockIdx.x * BN;

    // per-thread staging chunks: 2 for A tile, 2 for B tile (8 halves each)
    const int c0 = tid, c1 = tid + 256;            // 0..511
    const int r0 = c0 >> 2, g0 = (c0 & 3) * 8;     // row 0..127, colgrp
    const int r1 = c1 >> 2, g1 = (c1 & 3) * 8;
    const _Float16* gA0 = A  + (size_t)(rowBase + r0) * K + g0;
    const _Float16* gA1 = A  + (size_t)(rowBase + r1) * K + g1;
    const _Float16* gB0 = Wt + (size_t)(colBase + r0) * K + g0;
    const _Float16* gB1 = Wt + (size_t)(colBase + r1) * K + g1;

    unsigned aoff[2], a1off[2], boff[2], b1off[2];
#pragma unroll
    for (int b = 0; b < 2; ++b) {
        aoff[b]  = (unsigned)(uintptr_t)&smem[b][0][r0 * LDT + g0];
        a1off[b] = (unsigned)(uintptr_t)&smem[b][0][r1 * LDT + g1];
        boff[b]  = (unsigned)(uintptr_t)&smem[b][1][r0 * LDT + g0];
        b1off[b] = (unsigned)(uintptr_t)&smem[b][1][r1 * LDT + g1];
    }

    const int hl   = lane & 15;   // M row / N col within tile
    const int hi   = lane >> 4;   // K-group select per ISA 16-bit layout
    const int kgrp = hi * 8;

    v8f acc[4][2];
#pragma unroll
    for (int i = 0; i < 4; ++i)
#pragma unroll
        for (int j = 0; j < 2; ++j) acc[i][j] = {};

    // next-tile global pointers (advance BK halves per stage)
    const _Float16 *pA0 = gA0, *pA1 = gA1, *pB0 = gB0, *pB1 = gB1;

    auto stage = [&](int buf) {
        async_b128(aoff[buf],  pA0);
        async_b128(a1off[buf], pA1);
        async_b128(boff[buf],  pB0);
        async_b128(b1off[buf], pB1);
        pA0 += BK; pA1 += BK; pB0 += BK; pB1 += BK;
    };

    auto compute = [&](int buf) {
        const _Float16* As = smem[buf][0];
        const _Float16* Bs = smem[buf][1];
        v16h afrag[4], bfrag[2];
#pragma unroll
        for (int mi = 0; mi < 4; ++mi) {
            const _Float16* p = &As[(wm * 64 + mi * 16 + hl) * LDT + kgrp];
            v8h lo = *(const v8h*)(p);       // K = kgrp+0..7
            v8h hh = *(const v8h*)(p + 16);  // K = kgrp+16..23
            afrag[mi] = __builtin_shufflevector(lo, hh,
                0,1,2,3,4,5,6,7,8,9,10,11,12,13,14,15);
        }
#pragma unroll
        for (int ni = 0; ni < 2; ++ni) {
            const _Float16* p = &Bs[(wn * 32 + ni * 16 + hl) * LDT + kgrp];
            v8h lo = *(const v8h*)(p);
            v8h hh = *(const v8h*)(p + 16);
            bfrag[ni] = __builtin_shufflevector(lo, hh,
                0,1,2,3,4,5,6,7,8,9,10,11,12,13,14,15);
        }
#pragma unroll
        for (int mi = 0; mi < 4; ++mi)
#pragma unroll
            for (int ni = 0; ni < 2; ++ni)
                acc[mi][ni] = __builtin_amdgcn_wmma_f32_16x16x32_f16(
                    false, afrag[mi], false, bfrag[ni],
                    (short)0, acc[mi][ni], false, false);
    };

    const int nsteps = K / BK;    // even (10/32/64 for our shapes)

    stage(0);                     // tile 0 -> buf0
    // steady state: pairs of steps with statically-indexed buffers
    for (int it = 0; it < (nsteps - 2) / 2; ++it) {
        stage(1); wait_async_4(); __syncthreads();
        compute(0); __syncthreads();
        stage(0); wait_async_4(); __syncthreads();
        compute(1); __syncthreads();
    }
    // tail: last two steps
    stage(1); wait_async_4(); __syncthreads();
    compute(0); __syncthreads();
    wait_async_0(); __syncthreads();
    compute(1);

    // ---- epilogue: bias + relu, dual f32/f16 stores ----
#pragma unroll
    for (int mi = 0; mi < 4; ++mi) {
#pragma unroll
        for (int ni = 0; ni < 2; ++ni) {
            int col = colBase + wn * 32 + ni * 16 + hl;
            float bv = bias ? bias[col] : 0.0f;
#pragma unroll
            for (int r = 0; r < 8; ++r) {
                int row = rowBase + wm * 64 + mi * 16 + hi * 8 + r;
                float v = acc[mi][ni][r] + bv;
                if (relu) v = v > 0.0f ? v : 0.0f;
                if (outF) outF[(size_t)row * N + col] = v;
                if (outH) outH[(size_t)row * N + col] = (_Float16)v;
            }
        }
    }
}

// ------------- one-time weight transpose+convert: W[K,N]f32 -> Wt[N,K]f16 -------------
__global__ __launch_bounds__(256)
void transpose_f32_to_f16(const float* __restrict__ src, _Float16* __restrict__ dst,
                          int K, int N)
{
    __shared__ float tile[32][33];
    const int kb = blockIdx.y * 32, nb = blockIdx.x * 32;
    const int tx = threadIdx.x & 31, ty = threadIdx.x >> 5;  // 32x8
#pragma unroll
    for (int i = 0; i < 32; i += 8)
        tile[ty + i][tx] = src[(size_t)(kb + ty + i) * N + nb + tx];
    __syncthreads();
#pragma unroll
    for (int i = 0; i < 32; i += 8)
        dst[(size_t)(nb + ty + i) * K + kb + tx] = (_Float16)tile[tx][ty + i];
}

// ---------------- elementwise kernels ----------------
__global__ void init_carry(const float* __restrict__ pb, const float* __restrict__ pst,
                           float* __restrict__ belief, _Float16* __restrict__ belief16,
                           float* __restrict__ state)
{
    int i = blockIdx.x * blockDim.x + threadIdx.x;
    if (i < B_ * BEL_) { float v = pb[i]; belief[i] = v; belief16[i] = (_Float16)v; }
    if (i < B_ * STATE_) state[i] = pst[i];
}

__global__ void prep_sa(const float* __restrict__ state,
                        const float* __restrict__ actions_t,
                        const float* __restrict__ nonterm_t,
                        _Float16* __restrict__ sa)
{
    int i = blockIdx.x * blockDim.x + threadIdx.x;
    if (i >= B_ * (STATE_ + ACT_)) return;
    int b = i / (STATE_ + ACT_);
    int c = i - b * (STATE_ + ACT_);
    float v = (c < STATE_) ? state[b * STATE_ + c] * nonterm_t[b]
                           : actions_t[b * ACT_ + (c - STATE_)];
    sa[i] = (_Float16)v;
}

__global__ void gru_update(const float* __restrict__ gi, const float* __restrict__ gh,
                           const float* __restrict__ obs_t,
                           float* __restrict__ belief, _Float16* __restrict__ belief16,
                           _Float16* __restrict__ pos_in,
                           float* __restrict__ beliefs_out_t)
{
    int i = blockIdx.x * blockDim.x + threadIdx.x;
    if (i >= B_ * BEL_) return;
    int b = i >> 10, j = i & 1023;
    const float* gib = gi + (size_t)b * (3 * BEL_);
    const float* ghb = gh + (size_t)b * (3 * BEL_);
    float r = 1.0f / (1.0f + __expf(-(gib[j]        + ghb[j])));
    float z = 1.0f / (1.0f + __expf(-(gib[BEL_ + j] + ghb[BEL_ + j])));
    float n = tanhf(gib[2 * BEL_ + j] + r * ghb[2 * BEL_ + j]);
    float nb = (1.0f - z) * n + z * belief[i];
    belief[i] = nb;
    belief16[i] = (_Float16)nb;
    beliefs_out_t[i] = nb;
    pos_in[(size_t)b * (BEL_ + EMB_) + j]        = (_Float16)nb;
    pos_in[(size_t)b * (BEL_ + EMB_) + BEL_ + j] = (_Float16)obs_t[i];
}

__global__ void dist_head(const float* __restrict__ gout, const float* __restrict__ noise_t,
                          float* __restrict__ s_out, float* __restrict__ m_out,
                          float* __restrict__ std_out, float* __restrict__ state_next)
{
    int i = blockIdx.x * blockDim.x + threadIdx.x;
    if (i >= B_ * STATE_) return;
    int b = i >> 8, j = i & 255;
    float mean = gout[(size_t)b * (2 * STATE_) + j];
    float sr   = gout[(size_t)b * (2 * STATE_) + STATE_ + j];
    float sp   = (sr > 20.0f) ? sr : log1pf(__expf(sr));
    float std  = sp + 0.1f;
    float s    = mean + std * noise_t[i];
    s_out[i] = s; m_out[i] = mean; std_out[i] = std;
    if (state_next) state_next[i] = s;
}

// ---------------- host orchestration ----------------
extern "C" void kernel_launch(void* const* d_in, const int* in_sizes, int n_in,
                              void* d_out, int out_size, void* d_ws, size_t ws_size,
                              hipStream_t stream)
{
    (void)in_sizes; (void)n_in; (void)out_size; (void)ws_size;
    const float* prev_state  = (const float*)d_in[0];
    const float* actions     = (const float*)d_in[1];
    const float* prev_belief = (const float*)d_in[2];
    const float* obses       = (const float*)d_in[3];
    const float* nonterms    = (const float*)d_in[4];
    const float* noise_pri   = (const float*)d_in[5];
    const float* noise_pos   = (const float*)d_in[6];
    const float* enc_w = (const float*)d_in[7];  const float* enc_b = (const float*)d_in[8];
    const float* gru_wi = (const float*)d_in[9]; const float* gru_wh = (const float*)d_in[10];
    const float* gru_bi = (const float*)d_in[11];const float* gru_bh = (const float*)d_in[12];
    const float* pri_w1 = (const float*)d_in[13];const float* pri_b1 = (const float*)d_in[14];
    const float* pri_w2 = (const float*)d_in[15];const float* pri_b2 = (const float*)d_in[16];
    const float* pos_w1 = (const float*)d_in[17];const float* pos_b1 = (const float*)d_in[18];
    const float* pos_w2 = (const float*)d_in[19];const float* pos_b2 = (const float*)d_in[20];

    float* out = (float*)d_out;
    const size_t TB = (size_t)T_ * B_;
    float* o_beliefs = out;
    float* o_ps   = o_beliefs + TB * BEL_;
    float* o_pm   = o_ps   + TB * STATE_;
    float* o_pstd = o_pm   + TB * STATE_;
    float* o_qs   = o_pstd + TB * STATE_;
    float* o_qm   = o_qs   + TB * STATE_;
    float* o_qstd = o_qm   + TB * STATE_;

    char* wsb = (char*)d_ws;
    size_t off = 0;
    auto alloc = [&](size_t bytes) -> void* {
        off = (off + 255) & ~(size_t)255;
        void* p = wsb + off; off += bytes; return p;
    };
    const int SA = STATE_ + ACT_;
    _Float16* w16_enc = (_Float16*)alloc(sizeof(_Float16) * SA * HID_);
    _Float16* w16_gi  = (_Float16*)alloc(sizeof(_Float16) * HID_ * 3 * BEL_);
    _Float16* w16_gh  = (_Float16*)alloc(sizeof(_Float16) * BEL_ * 3 * BEL_);
    _Float16* w16_p1  = (_Float16*)alloc(sizeof(_Float16) * BEL_ * HID_);
    _Float16* w16_p2  = (_Float16*)alloc(sizeof(_Float16) * HID_ * 2 * STATE_);
    _Float16* w16_q1  = (_Float16*)alloc(sizeof(_Float16) * (BEL_ + EMB_) * HID_);
    _Float16* w16_q2  = (_Float16*)alloc(sizeof(_Float16) * HID_ * 2 * STATE_);
    float*    belief   = (float*)   alloc(sizeof(float)    * B_ * BEL_);
    _Float16* belief16 = (_Float16*)alloc(sizeof(_Float16) * B_ * BEL_);
    float*    state    = (float*)   alloc(sizeof(float)    * B_ * STATE_);
    _Float16* sa16     = (_Float16*)alloc(sizeof(_Float16) * B_ * SA);
    _Float16* enc16    = (_Float16*)alloc(sizeof(_Float16) * B_ * HID_);
    float*    gi       = (float*)   alloc(sizeof(float)    * B_ * 3 * BEL_);
    float*    gh       = (float*)   alloc(sizeof(float)    * B_ * 3 * BEL_);
    _Float16* h1_16    = (_Float16*)alloc(sizeof(_Float16) * B_ * HID_);
    _Float16* pos_in16 = (_Float16*)alloc(sizeof(_Float16) * B_ * (BEL_ + EMB_));
    _Float16* h2_16    = (_Float16*)alloc(sizeof(_Float16) * B_ * HID_);
    float*    pri_out  = (float*)   alloc(sizeof(float)    * B_ * 2 * STATE_);
    float*    pos_out  = (float*)   alloc(sizeof(float)    * B_ * 2 * STATE_);

    // ---- one-time per call: transpose + convert weights to f16 [N,K] ----
    auto convT = [&](const float* s, _Float16* d, int K, int N) {
        dim3 grid(N / 32, K / 32);
        transpose_f32_to_f16<<<grid, 256, 0, stream>>>(s, d, K, N);
    };
    convT(enc_w,  w16_enc, SA,          HID_);
    convT(gru_wi, w16_gi,  HID_,        3 * BEL_);
    convT(gru_wh, w16_gh,  BEL_,        3 * BEL_);
    convT(pri_w1, w16_p1,  BEL_,        HID_);
    convT(pri_w2, w16_p2,  HID_,        2 * STATE_);
    convT(pos_w1, w16_q1,  BEL_ + EMB_, HID_);
    convT(pos_w2, w16_q2,  HID_,        2 * STATE_);

    init_carry<<<(B_ * BEL_ + 255) / 256, 256, 0, stream>>>(
        prev_belief, prev_state, belief, belief16, state);

    auto gemm = [&](const _Float16* A, const _Float16* Wt, const float* bias,
                    float* oF, _Float16* oH, int M, int K, int N, int relu) {
        dim3 grid(N / BN, M / BM);
        gemm_f16_wmma<<<grid, 256, 0, stream>>>(A, Wt, bias, oF, oH, M, K, N, relu);
    };

    for (int t = 0; t < T_; ++t) {
        const float* act_t  = actions   + (size_t)t * B_ * ACT_;
        const float* obs_t  = obses     + (size_t)t * B_ * EMB_;
        const float* nt_t   = nonterms  + (size_t)t * B_;
        const float* npri_t = noise_pri + (size_t)t * B_ * STATE_;
        const float* npos_t = noise_pos + (size_t)t * B_ * STATE_;

        prep_sa<<<(B_ * SA + 255) / 256, 256, 0, stream>>>(state, act_t, nt_t, sa16);

        gemm(sa16, w16_enc, enc_b, nullptr, enc16, B_, SA, HID_, 1);
        gemm(enc16,    w16_gi, gru_bi, gi, nullptr, B_, HID_, 3 * BEL_, 0);
        gemm(belief16, w16_gh, gru_bh, gh, nullptr, B_, BEL_, 3 * BEL_, 0);

        gru_update<<<(B_ * BEL_ + 255) / 256, 256, 0, stream>>>(
            gi, gh, obs_t, belief, belief16, pos_in16,
            o_beliefs + (size_t)t * B_ * BEL_);

        gemm(belief16, w16_p1, pri_b1, nullptr, h1_16, B_, BEL_, HID_, 1);
        gemm(h1_16,    w16_p2, pri_b2, pri_out, nullptr, B_, HID_, 2 * STATE_, 0);
        dist_head<<<(B_ * STATE_ + 255) / 256, 256, 0, stream>>>(
            pri_out, npri_t,
            o_ps   + (size_t)t * B_ * STATE_,
            o_pm   + (size_t)t * B_ * STATE_,
            o_pstd + (size_t)t * B_ * STATE_, nullptr);

        gemm(pos_in16, w16_q1, pos_b1, nullptr, h2_16, B_, BEL_ + EMB_, HID_, 1);
        gemm(h2_16,    w16_q2, pos_b2, pos_out, nullptr, B_, HID_, 2 * STATE_, 0);
        dist_head<<<(B_ * STATE_ + 255) / 256, 256, 0, stream>>>(
            pos_out, npos_t,
            o_qs   + (size_t)t * B_ * STATE_,
            o_qm   + (size_t)t * B_ * STATE_,
            o_qstd + (size_t)t * B_ * STATE_, state);
    }
}
